// SAM2MultiScaleAttention_3221225472681
// MI455X (gfx1250) — compile-verified, matrix operand order checked by reading, not run on hardware
//
#include <hip/hip_runtime.h>
#include <hip/hip_bf16.h>
#include <math.h>

// ---------------------------------------------------------------------------
// SAM2 multi-scale attention for MI455X (gfx1250, wave32, WMMA)
//   hidden(4,64,64,192) -> qkv gemm -> pool(q) -> flash attention -> proj
// All matmuls run on v_wmma_f32_16x16x32_bf16 (bf16 in, f32 accumulate).
// ---------------------------------------------------------------------------

typedef __attribute__((ext_vector_type(16))) __bf16 v16bf;
typedef __attribute__((ext_vector_type(8)))  float  v8f;

union FragU { v16bf v; uint4 q[2]; };
union PackU { __bf16 h[8]; uint4 q; };

#define WMMA_BF16(A, B, C) \
  __builtin_amdgcn_wmma_f32_16x16x32_bf16(false, (A), false, (B), (short)0, (C), false, false)

// 16-bit A/B fragment loader for a row-major [row][K] bf16 matrix.
// CDNA5 16-bit 16x32 fragment: lane L holds row (L&15); halves 0..7 are
// K = kbase + (L>>4)*8 .. +7, halves 8..15 are that + 16.  => two b128 loads.
template <typename T>
__device__ __forceinline__ v16bf load_frag(const T* base, int ld, int row_base,
                                           int k_base, int lane) {
  const int r  = row_base + (lane & 15);
  const int k0 = k_base + ((lane >> 4) << 3);
  const T* p = base + (size_t)r * ld + k0;
  FragU f;
  f.q[0] = *reinterpret_cast<const uint4*>(p);
  f.q[1] = *reinterpret_cast<const uint4*>(p + 16);
  return f.v;
}

// ------------------------- elementwise prep kernels ------------------------

__global__ __launch_bounds__(256) void cvt_bf16_kernel(const float* __restrict__ in,
                                                       __bf16* __restrict__ out, int n) {
  int i = blockIdx.x * 256 + threadIdx.x;
  if (i < n) out[i] = (__bf16)in[i];
}

// out[c][r] = (bf16) in[r][c]   (weights -> [N][K] so B-frags load contiguous)
__global__ __launch_bounds__(256) void transpose_bf16_kernel(const float* __restrict__ in,
                                                             __bf16* __restrict__ out,
                                                             int R, int C) {
  int i = blockIdx.x * 256 + threadIdx.x;
  if (i < R * C) {
    int r = i / C, c = i - r * C;
    out[(size_t)c * R + r] = (__bf16)in[i];
  }
}

// 2x2 max pool of Q: [bh][4096][64] -> [bh][1024][64]
__global__ __launch_bounds__(256) void pool_q_kernel(const __bf16* __restrict__ Q,
                                                     __bf16* __restrict__ Qp) {
  int i = blockIdx.x * 256 + threadIdx.x;           // 24*1024*64 total
  if (i >= 24 * 1024 * 64) return;
  int d  = i & 63;
  int q  = (i >> 6) & 1023;
  int bh = i >> 16;
  int yq = q >> 5, xq = q & 31;
  const __bf16* base = Q + (size_t)bh * 4096 * 64;
  int t = (yq * 2) * 64 + xq * 2;                   // token = y*64 + x
  float m0 = (float)base[(size_t)(t)      * 64 + d];
  float m1 = (float)base[(size_t)(t + 1)  * 64 + d];
  float m2 = (float)base[(size_t)(t + 64) * 64 + d];
  float m3 = (float)base[(size_t)(t + 65) * 64 + d];
  Qp[((size_t)bh * 1024 + q) * 64 + d] = (__bf16)fmaxf(fmaxf(m0, m1), fmaxf(m2, m3));
}

// ------------------------------ QKV GEMM -----------------------------------
// X(16384x192) @ W(192x1152) + b.  Each wave: one 16(M) x 64(N) tile, K=192.
// N-tile of 64 == exactly one (which, head) pair; V is stored transposed so
// the C-fragment packs into a single b128 store per VGPR column group.

__global__ __launch_bounds__(256) void qkv_gemm_kernel(
    const __bf16* __restrict__ Xb, const __bf16* __restrict__ Wt,
    const float* __restrict__ bias,
    __bf16* __restrict__ Qf, __bf16* __restrict__ Kf, __bf16* __restrict__ Vt) {
  const int wave = (blockIdx.x * 256 + threadIdx.x) >> 5;
  const int lane = threadIdx.x & 31;
  const int mt = wave / 18;            // 1024 M-tiles of 16 rows
  const int nt = wave - mt * 18;       // 18 N-tiles of 64 cols
  const int l16 = lane & 15, hi = lane >> 4;

  v8f acc[4] = {};
  #pragma unroll
  for (int k = 0; k < 192; k += 32) {
    v16bf af = load_frag(Xb, 192, mt * 16, k, lane);
    #pragma unroll
    for (int v = 0; v < 4; ++v) {
      v16bf bf = load_frag(Wt, 192, nt * 64 + v * 16, k, lane);
      acc[v] = WMMA_BF16(af, bf, acc[v]);
    }
  }

  const int row0  = mt * 16;           // global token row
  const int b     = row0 >> 12;        // row0 / 4096
  const int hw0   = row0 & 4095;
  const int which = nt / 6;            // 0=q 1=k 2=v
  const int h     = nt - which * 6;
  const size_t bh = (size_t)b * 6 + h;

  #pragma unroll
  for (int v = 0; v < 4; ++v) {
    const int d  = v * 16 + l16;
    const float bv = bias[nt * 64 + v * 16 + l16];
    if (which == 2) {
      // V transposed: Vt[bh][d][token] -> 8 consecutive tokens per lane/VGPR
      PackU pk;
      #pragma unroll
      for (int r = 0; r < 8; ++r) pk.h[r] = (__bf16)(acc[v][r] + bv);
      *reinterpret_cast<uint4*>(Vt + (bh * 64 + d) * 4096 + hw0 + hi * 8) = pk.q;
    } else {
      __bf16* dst = (which == 0) ? Qf : Kf;      // [bh][token][64]
      #pragma unroll
      for (int r = 0; r < 8; ++r)
        dst[(bh * 4096 + (size_t)(hw0 + hi * 8 + r)) * 64 + d] =
            (__bf16)(acc[v][r] + bv);
    }
  }
}

// --------------------------- flash attention -------------------------------
// One wave per (bh, 16-row Q tile): online softmax over 4096 keys in chunks
// of 32.  S = Q*K^T via 4 WMMAs/chunk, P bounced through per-wave LDS to the
// A-fragment layout, O += P*V via 4 WMMAs/chunk.

__global__ __launch_bounds__(256) void attn_kernel(
    const __bf16* __restrict__ Qp, const __bf16* __restrict__ Kf,
    const __bf16* __restrict__ Vt, __bf16* __restrict__ Ob) {
  __shared__ __align__(16) __bf16 Pl[8][16][32];   // per-wave P tile
  const int wave = (blockIdx.x * 256 + threadIdx.x) >> 5;
  const int wv   = threadIdx.x >> 5;
  const int lane = threadIdx.x & 31;
  const int l16 = lane & 15, hi = lane >> 4;

  const int bh = wave >> 6;                        // 0..23
  const int qt = wave & 63;                        // Q tile within (b,h)
  const int b  = bh / 6, h = bh - b * 6;

  const __bf16* Qb = Qp + (size_t)bh * 1024 * 64;
  const __bf16* Kb = Kf + (size_t)bh * 4096 * 64;
  const __bf16* Vb = Vt + (size_t)bh * 64 * 4096;

  const v16bf qf0 = load_frag(Qb, 64, qt * 16, 0, lane);
  const v16bf qf1 = load_frag(Qb, 64, qt * 16, 32, lane);

  const float cfac = 0.125f * 1.44269504088896340736f;   // scale * log2(e)

  float mrow[8], lrow[8];
  v8f O[4] = {};
  #pragma unroll
  for (int r = 0; r < 8; ++r) { mrow[r] = -__builtin_inff(); lrow[r] = 0.f; }

  for (int j = 0; j < 4096; j += 32) {
    // ---- S = Q * K^T for 32 keys (two 16x16 tiles) ----
    v8f S0 = {}, S1 = {};
    {
      v16bf kf = load_frag(Kb, 64, j, 0, lane);
      S0 = WMMA_BF16(qf0, kf, S0);
      kf = load_frag(Kb, 64, j, 32, lane);
      S0 = WMMA_BF16(qf1, kf, S0);
      kf = load_frag(Kb, 64, j + 16, 0, lane);
      S1 = WMMA_BF16(qf0, kf, S1);
      kf = load_frag(Kb, 64, j + 16, 32, lane);
      S1 = WMMA_BF16(qf1, kf, S1);
    }
    // ---- online softmax: per-row max/sum via 16-lane shuffle reductions ----
    float nm[8], fs[8];
    #pragma unroll
    for (int r = 0; r < 8; ++r) {
      float v = fmaxf(S0[r], S1[r]);
      v = fmaxf(v, __shfl_xor(v, 1, 32));
      v = fmaxf(v, __shfl_xor(v, 2, 32));
      v = fmaxf(v, __shfl_xor(v, 4, 32));
      v = fmaxf(v, __shfl_xor(v, 8, 32));
      nm[r] = fmaxf(mrow[r], v);
      fs[r] = __builtin_amdgcn_exp2f((mrow[r] - nm[r]) * cfac);
      mrow[r] = nm[r];
    }
    #pragma unroll
    for (int r = 0; r < 8; ++r) {
      float p0 = __builtin_amdgcn_exp2f((S0[r] - nm[r]) * cfac);
      float p1 = __builtin_amdgcn_exp2f((S1[r] - nm[r]) * cfac);
      Pl[wv][hi * 8 + r][l16]      = (__bf16)p0;    // C-layout -> row-major P
      Pl[wv][hi * 8 + r][16 + l16] = (__bf16)p1;
      float s = p0 + p1;
      s += __shfl_xor(s, 1, 32);
      s += __shfl_xor(s, 2, 32);
      s += __shfl_xor(s, 4, 32);
      s += __shfl_xor(s, 8, 32);
      lrow[r] = lrow[r] * fs[r] + s;
    }
    #pragma unroll
    for (int v = 0; v < 4; ++v)
      #pragma unroll
      for (int r = 0; r < 8; ++r)
        O[v][r] *= fs[r];

    // DS ops are in-order within a wave: the A-frag loads below see the
    // stores above without a barrier.  Keep the compiler from reordering.
    asm volatile("" ::: "memory");
    const v16bf pf = load_frag(&Pl[wv][0][0], 32, 0, 0, lane);
    asm volatile("" ::: "memory");

    // ---- O += P * V (V transposed: contiguous along tokens) ----
    #pragma unroll
    for (int v = 0; v < 4; ++v) {
      v16bf vf = load_frag(Vb, 4096, v * 16, j, lane);
      O[v] = WMMA_BF16(pf, vf, O[v]);
    }
  }

  #pragma unroll
  for (int r = 0; r < 8; ++r) lrow[r] = 1.0f / lrow[r];
  const size_t orow0 = (size_t)b * 1024 + qt * 16;   // Ob is [b*1024+q][384]
  #pragma unroll
  for (int v = 0; v < 4; ++v) {
    const int col = h * 64 + v * 16 + l16;
    #pragma unroll
    for (int r = 0; r < 8; ++r)
      Ob[(orow0 + hi * 8 + r) * 384 + col] = (__bf16)(O[v][r] * lrow[r]);
  }
}

// ------------------------------ projection ---------------------------------
// Y = Ob(4096x384) @ proj_w(384x384) + b, f32 output straight to d_out.

__global__ __launch_bounds__(256) void proj_gemm_kernel(
    const __bf16* __restrict__ Ob, const __bf16* __restrict__ Pt,
    const float* __restrict__ bias, float* __restrict__ out) {
  const int wave = (blockIdx.x * 256 + threadIdx.x) >> 5;
  const int lane = threadIdx.x & 31;
  const int mt = wave / 6;             // 256 M-tiles
  const int nt = wave - mt * 6;        // 6 N-tiles of 64
  const int l16 = lane & 15, hi = lane >> 4;

  v8f acc[4] = {};
  #pragma unroll 4
  for (int k = 0; k < 384; k += 32) {
    v16bf af = load_frag(Ob, 384, mt * 16, k, lane);
    #pragma unroll
    for (int v = 0; v < 4; ++v) {
      v16bf bf = load_frag(Pt, 384, nt * 64 + v * 16, k, lane);
      acc[v] = WMMA_BF16(af, bf, acc[v]);
    }
  }
  #pragma unroll
  for (int v = 0; v < 4; ++v) {
    const int col = nt * 64 + v * 16 + l16;
    const float bv = bias[col];
    #pragma unroll
    for (int r = 0; r < 8; ++r)
      out[(size_t)(mt * 16 + hi * 8 + r) * 384 + col] = acc[v][r] + bv;
  }
}

// ------------------------------- launcher ----------------------------------

extern "C" void kernel_launch(void* const* d_in, const int* in_sizes, int n_in,
                              void* d_out, int out_size, void* d_ws, size_t ws_size,
                              hipStream_t stream) {
  (void)in_sizes; (void)n_in; (void)out_size; (void)ws_size;
  const float* hs     = (const float*)d_in[0];   // (4,64,64,192)
  const float* qkv_w  = (const float*)d_in[1];   // (192,1152)
  const float* qkv_b  = (const float*)d_in[2];   // (1152,)
  const float* proj_w = (const float*)d_in[3];   // (384,384)
  const float* proj_b = (const float*)d_in[4];   // (384,)
  float* out = (float*)d_out;                    // (4,32,32,384)
  char*  ws  = (char*)d_ws;

  size_t off = 0;
  __bf16* Xb = (__bf16*)(ws + off); off += (size_t)16384 * 192 * 2;   // X bf16
  __bf16* Wt = (__bf16*)(ws + off); off += (size_t)1152 * 192 * 2;    // qkv_w^T
  __bf16* Pt = (__bf16*)(ws + off); off += (size_t)384 * 384 * 2;     // proj_w^T
  __bf16* Qf = (__bf16*)(ws + off); off += (size_t)24 * 4096 * 64 * 2;// Q [bh][t][d]
  __bf16* Kf = (__bf16*)(ws + off); off += (size_t)24 * 4096 * 64 * 2;// K [bh][t][d]
  __bf16* Vt = (__bf16*)(ws + off); off += (size_t)24 * 64 * 4096 * 2;// V [bh][d][t]
  __bf16* Qp = (__bf16*)(ws + off); off += (size_t)24 * 1024 * 64 * 2;// pooled Q
  __bf16* Ob = (__bf16*)(ws + off); off += (size_t)4096 * 384 * 2;    // attn out

  cvt_bf16_kernel<<<(16384 * 192 + 255) / 256, 256, 0, stream>>>(hs, Xb, 16384 * 192);
  transpose_bf16_kernel<<<(192 * 1152 + 255) / 256, 256, 0, stream>>>(qkv_w, Wt, 192, 1152);
  transpose_bf16_kernel<<<(384 * 384 + 255) / 256, 256, 0, stream>>>(proj_w, Pt, 384, 384);

  qkv_gemm_kernel<<<(1024 * 18) / 8, 256, 0, stream>>>(Xb, Wt, qkv_b, Qf, Kf, Vt);
  pool_q_kernel<<<(24 * 1024 * 64 + 255) / 256, 256, 0, stream>>>(Qf, Qp);
  attn_kernel<<<(24 * 64) / 8, 256, 0, stream>>>(Qp, Kf, Vt, Ob);
  proj_gemm_kernel<<<(256 * 6) / 8, 256, 0, stream>>>(Ob, Pt, proj_b, out);
}
